// graphGAT_83099027243462
// MI455X (gfx1250) — matrix operand order repaired and checked by reading, not compiled, not run on hardware
//
#include <hip/hip_runtime.h>
#include <hip/hip_bf16.h>
#include <math.h>

typedef __attribute__((ext_vector_type(16))) _Float16 v16h;
typedef __attribute__((ext_vector_type(8)))  float    v8f;
typedef int v4i_g __attribute__((vector_size(16)));   // matches builtin's <4 x i32> pointee

union FragH { v16h v; _Float16 e[16]; };
union FragF { v8f  v; float    e[8];  };

#define T_TW 10000
#define SEQL 30
#define DIM  100
#define DP   128
#define G3   300
#define G3P  320
#define NUSR 40000
#define NNOD 50000
#define NEDG 320000
#define ETOT (NEDG + NNOD)
#define H1   8
#define C1   64
#define F1   512
#define C2   100
#define C2P  112

// -------- async copy to LDS (gfx1250) with compile-safe fallback --------
#ifdef __has_builtin
#if __has_builtin(__builtin_amdgcn_global_load_async_to_lds_b128)
#define HAVE_ASYNC_LDS 1
#endif
#endif
#ifndef HAVE_ASYNC_LDS
#define HAVE_ASYNC_LDS 0
#endif

__device__ inline void async_copy16(const void* g, void* l) {
#if HAVE_ASYNC_LDS
  void* gv = (void*)g;  // drop const; builtin takes non-const pointers
  __builtin_amdgcn_global_load_async_to_lds_b128(
      (__attribute__((address_space(1))) v4i_g*)gv,
      (__attribute__((address_space(3))) v4i_g*)l, 0, 0);
#else
  *(uint4*)l = *(const uint4*)g;
#endif
}

__device__ inline void wait_async_lds() {
#if HAVE_ASYNC_LDS
#ifdef __has_builtin
#if __has_builtin(__builtin_amdgcn_s_wait_asynccnt)
  __builtin_amdgcn_s_wait_asynccnt(0);
#else
  asm volatile("s_wait_asynccnt 0" ::: "memory");
#endif
#else
  asm volatile("s_wait_asynccnt 0" ::: "memory");
#endif
#endif
}

// ---------------- WMMA fragment helpers (CDNA5 16x16x32 f16) ----------------

__device__ inline v16h load_fragA(const _Float16* __restrict__ base, int lda) {
  // 16-bit A 16x32: lanes 0-15 row m hold K 0..7 & 16..23, lanes 16-31 hold K 8..15 & 24..31
  int lane = threadIdx.x & 31;
  int m = lane & 15;
  const _Float16* p = base + (size_t)m * lda + ((lane & 16) ? 8 : 0);
  FragH f;
#pragma unroll
  for (int i = 0; i < 8; ++i) { f.e[i] = p[i]; f.e[8 + i] = p[16 + i]; }
  return f.v;
}

__device__ inline v16h load_fragB(const _Float16* __restrict__ Bt, int ldb, int jtile, int k0) {
  // B 32x16 column n per lane: lanes 0-15 K 0..7 & 16..23, lanes 16-31 K 8..15 & 24..31 (Bt stored [n][k])
  int lane = threadIdx.x & 31;
  int n = lane & 15;
  const _Float16* p = Bt + (size_t)(jtile * 16 + n) * ldb + k0 + ((lane & 16) ? 8 : 0);
  FragH f;
#pragma unroll
  for (int i = 0; i < 8; ++i) { f.e[i] = p[i]; f.e[8 + i] = p[16 + i]; }
  return f.v;
}

__device__ inline v8f wmma16(v16h a, v16h b, v8f c) {
  return __builtin_amdgcn_wmma_f32_16x16x32_f16(false, a, false, b, (short)0, c, false, false);
}

// ---------------- prep kernels ----------------

__global__ void pad_weight(const float* __restrict__ src, _Float16* __restrict__ dst,
                           int srcR, int srcC, int prows, int pcols, int trans) {
  size_t i = (size_t)blockIdx.x * blockDim.x + threadIdx.x;
  size_t tot = (size_t)prows * pcols;
  if (i >= tot) return;
  int n = (int)(i / pcols), k = (int)(i % pcols);
  float v = 0.f;
  if (!trans) { if (n < srcR && k < srcC) v = src[(size_t)n * srcC + k]; }
  else        { if (k < srcR && n < srcC) v = src[(size_t)k * srcC + n]; }
  dst[i] = (_Float16)v;
}

__global__ void h_init(const float* __restrict__ h0, float* __restrict__ Hf,
                       _Float16* __restrict__ Hh) {
  size_t i = (size_t)blockIdx.x * blockDim.x + threadIdx.x;
  if (i >= (size_t)T_TW * DP) return;
  int t = (int)(i / DP), c = (int)(i % DP);
  float v = (c < DIM) ? h0[(size_t)t * DIM + c] : 0.f;
  if (c < DIM) Hf[(size_t)t * DIM + c] = v;
  Hh[i] = (_Float16)v;
}

__global__ void embed_tokens(const int* __restrict__ toks, const float* __restrict__ emb,
                             _Float16* __restrict__ Xh) {
  size_t i = (size_t)blockIdx.x * blockDim.x + threadIdx.x;
  if (i >= (size_t)SEQL * T_TW * DP) return;
  int c = (int)(i % DP);
  size_t st = i / DP;
  int t = (int)(st % T_TW);
  int s = (int)(st / T_TW);
  float v = 0.f;
  if (c < DIM) {
    int tok = toks[(size_t)t * SEQL + s];
    v = emb[(size_t)tok * DIM + c];
  }
  Xh[i] = (_Float16)v;
}

__global__ void user_fc1(const float* __restrict__ feats, const float* __restrict__ w1,
                         const float* __restrict__ b1, _Float16* __restrict__ Uh) {
  size_t i = (size_t)blockIdx.x * blockDim.x + threadIdx.x;
  if (i >= (size_t)NUSR * DP) return;
  int u = (int)(i / DP), j = (int)(i % DP);
  float v = 0.f;
  if (j < DIM) {
    float s = b1[j];
#pragma unroll
    for (int k = 0; k < 9; ++k) s += feats[(size_t)u * 9 + k] * w1[j * 9 + k];
    v = fmaxf(s, 0.f);
  }
  Uh[i] = (_Float16)v;
}

__global__ void concat_rows(const float* __restrict__ hn, _Float16* __restrict__ Xin) {
  size_t i = (size_t)blockIdx.x * blockDim.x + threadIdx.x;
  if (i >= (size_t)T_TW * DP) return;
  int t = (int)(i / DP), c = (int)(i % DP);
  size_t node = (t < 128) ? (size_t)t : (size_t)(NUSR + t);
  float v = (c < DIM) ? hn[(size_t)t * DIM + c] : 0.f;
  Xin[node * DP + c] = (_Float16)v;
}

__global__ void fill_f32(float* __restrict__ p, float v, size_t n) {
  size_t i = (size_t)blockIdx.x * blockDim.x + threadIdx.x;
  if (i < n) p[i] = v;
}
__global__ void fill_u32(unsigned* __restrict__ p, unsigned v, size_t n) {
  size_t i = (size_t)blockIdx.x * blockDim.x + threadIdx.x;
  if (i < n) p[i] = v;
}

// ---------------- generic WMMA GEMM: C(fp16) = A(fp16 pad) x Bt + bias ----------------

__global__ __launch_bounds__(256) void gemm_wmma_f16(
    const _Float16* __restrict__ A, int lda,
    const _Float16* __restrict__ Bt, int ldb,
    int ktiles, int ntiles,
    _Float16* __restrict__ C, int ldc,
    const float* __restrict__ bias, int bias_n) {
  int wave = threadIdx.x >> 5;
  int lane = threadIdx.x & 31;
  size_t m0 = (size_t)blockIdx.x * 16;
  const _Float16* Arow = A + m0 * lda;
  for (int j = wave; j < ntiles; j += 8) {
    v8f acc = {};
    for (int kk = 0; kk < ktiles; ++kk) {
      __builtin_prefetch(Arow + (kk + 1) * 32, 0, 1);
      v16h a = load_fragA(Arow + kk * 32, lda);
      v16h b = load_fragB(Bt, ldb, j, kk * 32);
      acc = wmma16(a, b, acc);
    }
    FragF r; r.v = acc;
    int n  = lane & 15;
    int mb = (lane & 16) ? 8 : 0;
    int col = j * 16 + n;
    float bv = (bias != nullptr && col < bias_n) ? bias[col] : 0.f;
#pragma unroll
    for (int i = 0; i < 8; ++i)
      C[(m0 + mb + i) * (size_t)ldc + col] = (_Float16)(r.e[i] + bv);
  }
}

// ---------------- fused GRU step (torch gate order r,z,n) ----------------
// A-tiles (x and h, 16x128 fp16 = 4KB each) are staged into LDS with
// global_load_async_to_lds_b128; WMMA A fragments are then read from LDS
// while B (weight) fragments stream from global/L2.

__global__ __launch_bounds__(256) void gru_step(
    const _Float16* __restrict__ X,    // [T][128] step input (fp16, padded)
    const _Float16* __restrict__ Hh,   // [T][128]
    const float*    __restrict__ Hf,   // [T][100]
    const _Float16* __restrict__ WiT,  // [320][128]
    const _Float16* __restrict__ WhT,  // [320][128]
    const float* __restrict__ bi, const float* __restrict__ bh,
    float* __restrict__ HfOut, _Float16* __restrict__ HhOut,
    _Float16* __restrict__ YsOut)      // nullable [T][128]
{
  __shared__ float gi_s[16][G3P];
  __shared__ float gh_s[16][G3P];
  __shared__ _Float16 xT[16 * DP];     // 4KB
  __shared__ _Float16 hT[16 * DP];     // 4KB
  int wave = threadIdx.x >> 5;
  int lane = threadIdx.x & 31;
  int tid  = threadIdx.x;
  size_t m0 = (size_t)blockIdx.x * 16;
  const _Float16* Ax = X  + m0 * DP;
  const _Float16* Ah = Hh + m0 * DP;

  // stage A tiles: 256 threads x 16B = 4KB per tile
  async_copy16((const char*)Ax + (size_t)tid * 16, (char*)xT + (size_t)tid * 16);
  async_copy16((const char*)Ah + (size_t)tid * 16, (char*)hT + (size_t)tid * 16);
  wait_async_lds();
  __syncthreads();

  for (int j = wave; j < G3P / 16; j += 8) {
    v8f ai = {}, ah = {};
    for (int kk = 0; kk < DP / 32; ++kk) {
      v16h ax  = load_fragA(xT + kk * 32, DP);
      v16h axh = load_fragA(hT + kk * 32, DP);
      v16h bfi = load_fragB(WiT, DP, j, kk * 32);
      v16h bfh = load_fragB(WhT, DP, j, kk * 32);
      ai = wmma16(ax,  bfi, ai);
      ah = wmma16(axh, bfh, ah);
    }
    int n  = lane & 15;
    int mb = (lane & 16) ? 8 : 0;
    int col = j * 16 + n;
    float bvi = (col < G3) ? bi[col] : 0.f;
    float bvh = (col < G3) ? bh[col] : 0.f;
    FragF ri; ri.v = ai;
    FragF rh; rh.v = ah;
#pragma unroll
    for (int i = 0; i < 8; ++i) {
      gi_s[mb + i][col] = ri.e[i] + bvi;
      gh_s[mb + i][col] = rh.e[i] + bvh;
    }
  }
  __syncthreads();

  for (int idx = threadIdx.x; idx < 16 * DP; idx += 256) {
    int m = idx >> 7, c = idx & 127;
    size_t row = m0 + m;
    float hv = 0.f;
    if (c < DIM) {
      float r  = 1.f / (1.f + expf(-(gi_s[m][c]           + gh_s[m][c])));
      float z  = 1.f / (1.f + expf(-(gi_s[m][DIM + c]     + gh_s[m][DIM + c])));
      float nn = tanhf(gi_s[m][2 * DIM + c] + r * gh_s[m][2 * DIM + c]);
      float hold = Hf[row * DIM + c];
      hv = (1.f - z) * nn + z * hold;
      HfOut[row * DIM + c] = hv;
    }
    _Float16 h16 = (_Float16)hv;
    HhOut[row * DP + c] = h16;
    if (YsOut) YsOut[row * DP + c] = h16;
  }
}

// ---------------- GAT pieces ----------------

__global__ void gat_scores(const _Float16* __restrict__ xh, int ld, int heads, int ch,
                           const float* __restrict__ asrc, const float* __restrict__ adst,
                           float* __restrict__ ss, float* __restrict__ sd) {
  size_t i = (size_t)blockIdx.x * blockDim.x + threadIdx.x;
  if (i >= (size_t)NNOD * heads) return;
  int h = (int)(i % heads);
  size_t n = i / heads;
  const _Float16* p = xh + n * ld + h * ch;
  float s1 = 0.f, s2 = 0.f;
  for (int c = 0; c < ch; ++c) {
    float x = (float)p[c];
    s1 += x * asrc[h * ch + c];
    s2 += x * adst[h * ch + c];
  }
  ss[i] = s1; sd[i] = s2;
}

__device__ inline unsigned encf(float f) {
  unsigned u = __float_as_uint(f);
  return (u & 0x80000000u) ? ~u : (u | 0x80000000u);
}
__device__ inline float decf(unsigned u) {
  return __uint_as_float((u & 0x80000000u) ? (u & 0x7FFFFFFFu) : ~u);
}

__device__ inline void edge_sd(const int* __restrict__ ei, size_t e, int& s, int& d) {
  if (e < (size_t)NEDG) { s = ei[e]; d = ei[(size_t)NEDG + e]; }
  else                  { s = d = (int)(e - NEDG); }
}

__global__ void edge_max(const int* __restrict__ ei, int heads,
                         const float* __restrict__ ss, const float* __restrict__ sd,
                         unsigned* __restrict__ mx) {
  size_t i = (size_t)blockIdx.x * blockDim.x + threadIdx.x;
  if (i >= (size_t)ETOT * heads) return;
  int h = (int)(i % heads);
  size_t e = i / heads;
  int s, d; edge_sd(ei, e, s, d);
  float a = ss[(size_t)s * heads + h] + sd[(size_t)d * heads + h];
  a = (a > 0.f) ? a : 0.2f * a;
  atomicMax(&mx[(size_t)d * heads + h], encf(a));
}

__global__ void edge_expsum(const int* __restrict__ ei, int heads,
                            const float* __restrict__ ss, const float* __restrict__ sd,
                            const unsigned* __restrict__ mx,
                            float* __restrict__ ea, float* __restrict__ den) {
  size_t i = (size_t)blockIdx.x * blockDim.x + threadIdx.x;
  if (i >= (size_t)ETOT * heads) return;
  int h = (int)(i % heads);
  size_t e = i / heads;
  int s, d; edge_sd(ei, e, s, d);
  float a = ss[(size_t)s * heads + h] + sd[(size_t)d * heads + h];
  a = (a > 0.f) ? a : 0.2f * a;
  float m = decf(mx[(size_t)d * heads + h]);
  float v = expf(a - m);
  ea[i] = v;
  atomicAdd(&den[(size_t)d * heads + h], v);
}

__global__ void edge_agg(const int* __restrict__ ei, int heads, int ch, int ld,
                         const _Float16* __restrict__ xh, const float* __restrict__ ea,
                         const float* __restrict__ den, float* __restrict__ agg, int fdim) {
  size_t i = (size_t)blockIdx.x * blockDim.x + threadIdx.x;
  if (i >= (size_t)ETOT * heads) return;
  int h = (int)(i % heads);
  size_t e = i / heads;
  int s, d; edge_sd(ei, e, s, d);
  float w = ea[i] / (den[(size_t)d * heads + h] + 1e-16f);
  const _Float16* xp = xh + (size_t)s * ld + h * ch;
  float* op = agg + (size_t)d * fdim + h * ch;
  for (int c = 0; c < ch; ++c) atomicAdd(&op[c], w * (float)xp[c]);
}

__global__ void finish_relu_h(const float* __restrict__ agg, const float* __restrict__ b,
                              _Float16* __restrict__ outp, int fdim, size_t n) {
  size_t i = (size_t)blockIdx.x * blockDim.x + threadIdx.x;
  if (i >= n) return;
  outp[i] = (_Float16)fmaxf(agg[i] + b[i % fdim], 0.f);
}

__global__ void finish_out(const float* __restrict__ agg, const float* __restrict__ b,
                           float* __restrict__ outp) {
  size_t i = (size_t)blockIdx.x * blockDim.x + threadIdx.x;
  if (i >= (size_t)NNOD * C2) return;
  outp[i] = fmaxf(agg[i] + b[i % C2], 0.f);
}

// ---------------- host orchestration ----------------

extern "C" void kernel_launch(void* const* d_in, const int* in_sizes, int n_in,
                              void* d_out, int out_size, void* d_ws, size_t ws_size,
                              hipStream_t stream) {
  (void)in_sizes; (void)n_in; (void)out_size; (void)ws_size;
  const float* user_feats = (const float*)d_in[1];
  const int*   toks  = (const int*)d_in[2];
  const int*   ei    = (const int*)d_in[3];
  const float* h0    = (const float*)d_in[4];
  const float* emb   = (const float*)d_in[5];
  const float* fc_w1 = (const float*)d_in[6];
  const float* fc_b1 = (const float*)d_in[7];
  const float* fc_w2 = (const float*)d_in[8];
  const float* fc_b2 = (const float*)d_in[9];
  const float* wi0 = (const float*)d_in[10];
  const float* wh0 = (const float*)d_in[11];
  const float* bi0 = (const float*)d_in[12];
  const float* bh0 = (const float*)d_in[13];
  const float* wi1 = (const float*)d_in[14];
  const float* wh1 = (const float*)d_in[15];
  const float* bi1 = (const float*)d_in[16];
  const float* bh1 = (const float*)d_in[17];
  const float* g1_w    = (const float*)d_in[18];
  const float* g1_asrc = (const float*)d_in[19];
  const float* g1_adst = (const float*)d_in[20];
  const float* g1_b    = (const float*)d_in[21];
  const float* g2_w    = (const float*)d_in[22];
  const float* g2_asrc = (const float*)d_in[23];
  const float* g2_adst = (const float*)d_in[24];
  const float* g2_b    = (const float*)d_in[25];
  float* outp = (float*)d_out;

  char* ws = (char*)d_ws;
  size_t off = 0;
  auto alloc = [&](size_t bytes) -> char* {
    char* p = ws + off;
    off += (bytes + 255) & ~(size_t)255;
    return p;
  };

  _Float16* wi0t = (_Float16*)alloc((size_t)G3P * DP * 2);
  _Float16* wh0t = (_Float16*)alloc((size_t)G3P * DP * 2);
  _Float16* wi1t = (_Float16*)alloc((size_t)G3P * DP * 2);
  _Float16* wh1t = (_Float16*)alloc((size_t)G3P * DP * 2);
  _Float16* fc2t = (_Float16*)alloc((size_t)DP * DP * 2);
  _Float16* g1t  = (_Float16*)alloc((size_t)F1 * DP * 2);
  _Float16* g2t  = (_Float16*)alloc((size_t)C2P * F1 * 2);
  _Float16* Xh   = (_Float16*)alloc((size_t)SEQL * T_TW * DP * 2);
  _Float16* Ysh  = (_Float16*)alloc((size_t)SEQL * T_TW * DP * 2);
  _Float16* Hh_a = (_Float16*)alloc((size_t)T_TW * DP * 2);
  _Float16* Hh_b = (_Float16*)alloc((size_t)T_TW * DP * 2);
  float* Hf_a = (float*)alloc((size_t)T_TW * DIM * 4);
  float* Hf_b = (float*)alloc((size_t)T_TW * DIM * 4);
  _Float16* Uh  = (_Float16*)alloc((size_t)NUSR * DP * 2);
  _Float16* Xin = (_Float16*)alloc((size_t)NNOD * DP * 2);
  _Float16* XH1 = (_Float16*)alloc((size_t)NNOD * F1 * 2);
  float*    SS1 = (float*)alloc((size_t)NNOD * H1 * 4);
  float*    SD1 = (float*)alloc((size_t)NNOD * H1 * 4);
  unsigned* MX1 = (unsigned*)alloc((size_t)NNOD * H1 * 4);
  float*    DEN1 = (float*)alloc((size_t)NNOD * H1 * 4);
  float*    EA1  = (float*)alloc((size_t)ETOT * H1 * 4);
  float*    AGG1 = (float*)alloc((size_t)NNOD * F1 * 4);
  _Float16* X2h  = (_Float16*)alloc((size_t)NNOD * F1 * 2);
  _Float16* XH2  = (_Float16*)alloc((size_t)NNOD * C2P * 2);
  float*    SS2  = (float*)alloc((size_t)NNOD * 4);
  float*    SD2  = (float*)alloc((size_t)NNOD * 4);
  unsigned* MX2  = (unsigned*)alloc((size_t)NNOD * 4);
  float*    DEN2 = (float*)alloc((size_t)NNOD * 4);
  float*    EA2  = (float*)alloc((size_t)ETOT * 4);
  float*    AGG2 = (float*)alloc((size_t)NNOD * C2 * 4);

  dim3 blk(256);
  auto g1d = [](size_t n) { return dim3((unsigned)((n + 255) / 256)); };

  // --- weight prep (fp16 padded, Bt[n][k] layout) ---
  pad_weight<<<g1d((size_t)G3P * DP), blk, 0, stream>>>(wi0, wi0t, G3, DIM, G3P, DP, 0);
  pad_weight<<<g1d((size_t)G3P * DP), blk, 0, stream>>>(wh0, wh0t, G3, DIM, G3P, DP, 0);
  pad_weight<<<g1d((size_t)G3P * DP), blk, 0, stream>>>(wi1, wi1t, G3, DIM, G3P, DP, 0);
  pad_weight<<<g1d((size_t)G3P * DP), blk, 0, stream>>>(wh1, wh1t, G3, DIM, G3P, DP, 0);
  pad_weight<<<g1d((size_t)DP * DP), blk, 0, stream>>>(fc_w2, fc2t, DIM, DIM, DP, DP, 0);
  pad_weight<<<g1d((size_t)F1 * DP), blk, 0, stream>>>(g1_w, g1t, DIM, F1, F1, DP, 1);
  pad_weight<<<g1d((size_t)C2P * F1), blk, 0, stream>>>(g2_w, g2t, F1, C2, C2P, F1, 1);

  // --- token embeddings, user encoder ---
  embed_tokens<<<g1d((size_t)SEQL * T_TW * DP), blk, 0, stream>>>(toks, emb, Xh);
  user_fc1<<<g1d((size_t)NUSR * DP), blk, 0, stream>>>(user_feats, fc_w1, fc_b1, Uh);
  // fc2 writes user_embedding straight into the concat slot (rows 128..40127 of Xin)
  gemm_wmma_f16<<<dim3(NUSR / 16), blk, 0, stream>>>(
      Uh, DP, fc2t, DP, DP / 32, DP / 16, Xin + (size_t)128 * DP, DP, fc_b2, DIM);

  // --- GRU layer 0 ---
  h_init<<<g1d((size_t)T_TW * DP), blk, 0, stream>>>(h0, Hf_a, Hh_a);
  _Float16* hh_c = Hh_a; _Float16* hh_n = Hh_b;
  float* hf_c = Hf_a; float* hf_n = Hf_b;
  for (int s = 0; s < SEQL; ++s) {
    gru_step<<<dim3(T_TW / 16), blk, 0, stream>>>(
        Xh + (size_t)s * T_TW * DP, hh_c, hf_c, wi0t, wh0t, bi0, bh0,
        hf_n, hh_n, Ysh + (size_t)s * T_TW * DP);
    { _Float16* t = hh_c; hh_c = hh_n; hh_n = t; }
    { float* t = hf_c; hf_c = hf_n; hf_n = t; }
  }
  // --- GRU layer 1 ---
  h_init<<<g1d((size_t)T_TW * DP), blk, 0, stream>>>(h0 + (size_t)T_TW * DIM, Hf_a, Hh_a);
  hh_c = Hh_a; hh_n = Hh_b; hf_c = Hf_a; hf_n = Hf_b;
  for (int s = 0; s < SEQL; ++s) {
    gru_step<<<dim3(T_TW / 16), blk, 0, stream>>>(
        Ysh + (size_t)s * T_TW * DP, hh_c, hf_c, wi1t, wh1t, bi1, bh1,
        hf_n, hh_n, nullptr);
    { _Float16* t = hh_c; hh_c = hh_n; hh_n = t; }
    { float* t = hf_c; hf_c = hf_n; hf_n = t; }
  }
  const float* hn = hf_c;  // final hidden state [T,100]

  // --- concat x_input = [hn[:128], user_embedding, hn[128:]] (fp16 padded) ---
  concat_rows<<<g1d((size_t)T_TW * DP), blk, 0, stream>>>(hn, Xin);

  // --- GAT layer 1: xh = x @ W1 (50000x512) ---
  gemm_wmma_f16<<<dim3(NNOD / 16), blk, 0, stream>>>(
      Xin, DP, g1t, DP, DP / 32, F1 / 16, XH1, F1, nullptr, 0);
  gat_scores<<<g1d((size_t)NNOD * H1), blk, 0, stream>>>(XH1, F1, H1, C1, g1_asrc, g1_adst, SS1, SD1);
  fill_u32<<<g1d((size_t)NNOD * H1), blk, 0, stream>>>(MX1, 0u, (size_t)NNOD * H1);
  fill_f32<<<g1d((size_t)NNOD * H1), blk, 0, stream>>>(DEN1, 0.f, (size_t)NNOD * H1);
  fill_f32<<<g1d((size_t)NNOD * F1), blk, 0, stream>>>(AGG1, 0.f, (size_t)NNOD * F1);
  edge_max<<<g1d((size_t)ETOT * H1), blk, 0, stream>>>(ei, H1, SS1, SD1, MX1);
  edge_expsum<<<g1d((size_t)ETOT * H1), blk, 0, stream>>>(ei, H1, SS1, SD1, MX1, EA1, DEN1);
  edge_agg<<<g1d((size_t)ETOT * H1), blk, 0, stream>>>(ei, H1, C1, F1, XH1, EA1, DEN1, AGG1, F1);
  finish_relu_h<<<g1d((size_t)NNOD * F1), blk, 0, stream>>>(AGG1, g1_b, X2h, F1, (size_t)NNOD * F1);

  // --- GAT layer 2: xh = x @ W2 (50000x100, padded 112) ---
  gemm_wmma_f16<<<dim3(NNOD / 16), blk, 0, stream>>>(
      X2h, F1, g2t, F1, F1 / 32, C2P / 16, XH2, C2P, nullptr, 0);
  gat_scores<<<g1d((size_t)NNOD), blk, 0, stream>>>(XH2, C2P, 1, C2, g2_asrc, g2_adst, SS2, SD2);
  fill_u32<<<g1d((size_t)NNOD), blk, 0, stream>>>(MX2, 0u, (size_t)NNOD);
  fill_f32<<<g1d((size_t)NNOD), blk, 0, stream>>>(DEN2, 0.f, (size_t)NNOD);
  fill_f32<<<g1d((size_t)NNOD * C2), blk, 0, stream>>>(AGG2, 0.f, (size_t)NNOD * C2);
  edge_max<<<g1d((size_t)ETOT), blk, 0, stream>>>(ei, 1, SS2, SD2, MX2);
  edge_expsum<<<g1d((size_t)ETOT), blk, 0, stream>>>(ei, 1, SS2, SD2, MX2, EA2, DEN2);
  edge_agg<<<g1d((size_t)ETOT), blk, 0, stream>>>(ei, 1, C2, C2P, XH2, EA2, DEN2, AGG2, C2);
  finish_out<<<g1d((size_t)NNOD * C2), blk, 0, stream>>>(AGG2, g2_b, outp);
}
